// BDH_67765993996439
// MI455X (gfx1250) — matrix operand order repaired and checked by reading, not compiled, not run on hardware
//
#include <hip/hip_runtime.h>
#include <hip/hip_bf16.h>
#include <stdint.h>

typedef __attribute__((ext_vector_type(16))) __bf16 v16bf;
typedef __attribute__((ext_vector_type(8)))  __bf16 v8bf;
typedef __attribute__((ext_vector_type(8)))  float  v8f;

constexpr int   Dd   = 256;
constexpr int   NH   = 4;
constexpr int   Tt   = 512;
constexpr int   Bb   = 2;
constexpr int   Nn   = 8192;
constexpr int   BT   = Bb * Tt;     // 1024
constexpr int   NHN  = NH * Nn;     // 32768
constexpr float LNEPS = 1e-5f;
constexpr float TWOPI = 6.283185307179586f;

enum { EP_F32 = 0, EP_RELU_BF16 = 1, EP_GATE_BF16 = 2, EP_MASK_BF16 = 3 };

// ---------------------------------------------------------------------------
// gfx1250 async global->LDS copy (ASYNCcnt-tracked), 16 bytes per lane.
// ISA: GLOBAL_LOAD_ASYNC_TO_LDS_B128  vdst = per-lane LDS byte address,
//      vaddr = 64-bit global address (GV mode).
// ---------------------------------------------------------------------------
__device__ inline void g2l_b128(const void* g, void* l)
{
    uint32_t lo = (uint32_t)(uintptr_t)l;            // LDS offset (low 32 bits)
    unsigned long long ga = (unsigned long long)(uintptr_t)g;
    asm volatile("global_load_async_to_lds_b128 %0, %1, off"
                 :: "v"(lo), "v"(ga) : "memory");
}

__device__ inline void wait_async5()
{
    asm volatile("s_wait_asynccnt 0x5" ::: "memory");
}

__device__ inline void wait_async0()
{
    asm volatile("s_wait_asynccnt 0x0" ::: "memory");
}

// ---------------------------------------------------------------------------
// Generic NT bf16 WMMA GEMM:  C[m,n] = sum_k A[m*lda+k] * B[n*ldb+k]
// Block tile 64x256 (8 waves; wave tile 32x64 = 2x4 wmma tiles), BK = 32.
// Double-buffered async global->LDS staging; fragments read from LDS
// following the CDNA5 16-bit WMMA per-lane layouts:
//   A 16x32: lane holds row m=lane&15, K chunks [8*lh..+7] and [16+8*lh..+7]
//   B 32x16: lane holds row n=lane&15, K chunks [8*lh..+7] and [16+8*lh..+7]
// ---------------------------------------------------------------------------
template<int EPI>
__global__ __launch_bounds__(256) void k_gemm_nt(
    const __bf16* __restrict__ A, const __bf16* __restrict__ Bm,
    void* __restrict__ Cv, const __bf16* __restrict__ G,
    int K, int lda, int ldb, int ldc, int ldg, int zmod,
    long as1, long as2, long bs1, long bs2,
    long cs1, long cs2, long gs1, long gs2)
{
    __shared__ __bf16 lA[2][64][40];    // 32 used + 8 pad halfs per row
    __shared__ __bf16 lB[2][256][40];

    const int z  = blockIdx.z;
    const int z1 = z / zmod, z2 = z % zmod;
    A  += z1 * as1 + z2 * as2;
    Bm += z1 * bs1 + z2 * bs2;
    const long coff = z1 * cs1 + z2 * cs2;
    const long goff = z1 * gs1 + z2 * gs2;

    const int tid   = threadIdx.x;
    const int lane  = tid & 31;
    const int wave  = tid >> 5;
    const int wm    = wave >> 2;            // 0..1
    const int wn    = wave & 3;             // 0..3
    const int mblk  = blockIdx.y * 64;
    const int nblk  = blockIdx.x * 256;
    const int lm    = lane & 15;
    const int lh    = lane >> 4;

    // staging coords: each thread copies one 16B chunk of A, four of B
    const int srow = tid >> 2;              // 0..63
    const int schk = (tid & 3) * 8;         // 0,8,16,24

    auto issue_tile = [&](int k0, int buf) {
        g2l_b128(A + (size_t)(mblk + srow) * lda + k0 + schk,
                 &lA[buf][srow][schk]);
#pragma unroll
        for (int q = 0; q < 4; ++q)
            g2l_b128(Bm + (size_t)(nblk + q * 64 + srow) * ldb + k0 + schk,
                     &lB[buf][q * 64 + srow][schk]);
    };

    v8f acc[2][4];
#pragma unroll
    for (int i = 0; i < 2; ++i)
#pragma unroll
        for (int j = 0; j < 4; ++j)
#pragma unroll
            for (int e = 0; e < 8; ++e) acc[i][j][e] = 0.f;

    issue_tile(0, 0);
    int buf = 0;
    for (int k0 = 0; k0 < K; k0 += 32) {
        if (k0 + 32 < K) {
            issue_tile(k0 + 32, buf ^ 1);
            wait_async5();                  // in-order: current tile landed
        } else {
            wait_async0();
        }
        __syncthreads();                    // tile visible to all waves

        v16bf af[2];
#pragma unroll
        for (int i = 0; i < 2; ++i) {
            const __bf16* p = &lA[buf][wm * 32 + i * 16 + lm][8 * lh];
            v8bf c0 = *(const v8bf*)(p);
            v8bf c1 = *(const v8bf*)(p + 16);
#pragma unroll
            for (int e = 0; e < 8; ++e) { af[i][e] = c0[e]; af[i][e + 8] = c1[e]; }
        }
        v16bf bfr[4];
#pragma unroll
        for (int j = 0; j < 4; ++j) {
            const __bf16* p = &lB[buf][wn * 64 + j * 16 + lm][8 * lh];
            v8bf b0 = *(const v8bf*)(p);
            v8bf b1 = *(const v8bf*)(p + 16);
#pragma unroll
            for (int e = 0; e < 8; ++e) { bfr[j][e] = b0[e]; bfr[j][e + 8] = b1[e]; }
        }
#pragma unroll
        for (int i = 0; i < 2; ++i)
#pragma unroll
            for (int j = 0; j < 4; ++j)
                acc[i][j] = __builtin_amdgcn_wmma_f32_16x16x32_bf16(
                    false, af[i], false, bfr[j], (short)0, acc[i][j], false, false);

        __syncthreads();                    // done reading buf before overwrite
        buf ^= 1;
    }

    // C/D layout: vgpr r -> row = mbase+i*16+8*lh+r, col = nbase+j*16+lm
#pragma unroll
    for (int i = 0; i < 2; ++i) {
#pragma unroll
        for (int j = 0; j < 4; ++j) {
            const int col   = nblk + wn * 64 + j * 16 + lm;
            const int rbase = mblk + wm * 32 + i * 16 + 8 * lh;
#pragma unroll
            for (int r = 0; r < 8; ++r) {
                const int row = rbase + r;
                float v = acc[i][j][r];
                if constexpr (EPI == EP_F32) {
                    float* C = (float*)Cv + coff;
                    C[(size_t)row * ldc + col] = v;
                } else if constexpr (EPI == EP_RELU_BF16) {
                    __bf16* C = (__bf16*)Cv + coff;
                    C[(size_t)row * ldc + col] = (__bf16)fmaxf(v, 0.f);
                } else if constexpr (EPI == EP_GATE_BF16) {
                    __bf16* C = (__bf16*)Cv + coff;
                    float g = (float)G[(size_t)row * ldg + col + goff];
                    C[(size_t)row * ldc + col] = (__bf16)(fmaxf(v, 0.f) * g);
                } else { // EP_MASK_BF16: strict lower-triangular causal mask
                    __bf16* C = (__bf16*)Cv + coff;
                    C[(size_t)row * ldc + col] = (col < row) ? (__bf16)v : (__bf16)0.f;
                }
            }
        }
    }
}

// ---------------------------------------------------------------------------
// fp32 (R,C) -> bf16 transposed (C,R)
// ---------------------------------------------------------------------------
__global__ __launch_bounds__(256) void k_w_transpose(
    const float* __restrict__ src, __bf16* __restrict__ dst,
    int R, int C, long ss, long ds)
{
    src += (size_t)blockIdx.z * ss;
    dst += (size_t)blockIdx.z * ds;
    size_t i = (size_t)blockIdx.x * 256 + threadIdx.x;
    size_t r = i / (size_t)C, c = i % (size_t)C;
    dst[c * (size_t)R + r] = (__bf16)src[i];
}

// ---------------------------------------------------------------------------
__device__ inline float blk_sum(float v)
{
    __shared__ float sh[256];
    int t = threadIdx.x;
    sh[t] = v; __syncthreads();
    for (int s = 128; s > 0; s >>= 1) {
        if (t < s) sh[t] += sh[t + s];
        __syncthreads();
    }
    float r = sh[0]; __syncthreads();
    return r;
}

__global__ __launch_bounds__(256) void k_embed_ln(
    const int* __restrict__ idx, const float* __restrict__ embed,
    float* __restrict__ x, __bf16* __restrict__ xbf, __bf16* __restrict__ xT)
{
    int row = blockIdx.x, d = threadIdx.x;
    int tok = idx[row];
    float v  = embed[(size_t)tok * Dd + d];
    float m  = blk_sum(v) * (1.f / Dd);
    float dv = v - m;
    float var = blk_sum(dv * dv) * (1.f / Dd);
    float o  = dv * rsqrtf(var + LNEPS);
    x[(size_t)row * Dd + d]   = o;
    xbf[(size_t)row * Dd + d] = (__bf16)o;
    int b = row / Tt, t = row % Tt;
    xT[((size_t)b * Dd + d) * Tt + t] = (__bf16)o;
}

__global__ __launch_bounds__(256) void k_ln_rows(
    const float* __restrict__ in, __bf16* __restrict__ out)
{
    int row = blockIdx.x, d = threadIdx.x;
    float v  = in[(size_t)row * Dd + d];
    float m  = blk_sum(v) * (1.f / Dd);
    float dv = v - m;
    float var = blk_sum(dv * dv) * (1.f / Dd);
    out[(size_t)row * Dd + d] = (__bf16)(dv * rsqrtf(var + LNEPS));
}

__global__ __launch_bounds__(256) void k_resid_ln(
    float* __restrict__ x, const float* __restrict__ ymlp,
    __bf16* __restrict__ xbf, __bf16* __restrict__ xT)
{
    int row = blockIdx.x, d = threadIdx.x;
    float v  = x[(size_t)row * Dd + d] + ymlp[(size_t)row * Dd + d];
    float m  = blk_sum(v) * (1.f / Dd);
    float dv = v - m;
    float var = blk_sum(dv * dv) * (1.f / Dd);
    float o  = dv * rsqrtf(var + LNEPS);
    x[(size_t)row * Dd + d]   = o;
    xbf[(size_t)row * Dd + d] = (__bf16)o;
    int b = row / Tt, t = row % Tt;
    xT[((size_t)b * Dd + d) * Tt + t] = (__bf16)o;
}

// ---------------------------------------------------------------------------
// RoPE on x_sparse (layout B,T,NH,N), pairwise rotation; same angle per pair.
// freq(n) = theta^(-floor(n/2)*2/N) / 2pi,  theta = 2^16
// ---------------------------------------------------------------------------
__global__ __launch_bounds__(256) void k_rope(
    const __bf16* __restrict__ xs, __bf16* __restrict__ qr)
{
    size_t pid = (size_t)blockIdx.x * 256 + threadIdx.x;  // pair id
    int    np  = (int)(pid % (Nn / 2));
    size_t q   = pid / (Nn / 2);
    int    h   = (int)(q % NH);  q /= NH;
    int    t   = (int)(q % Tt);
    int    b   = (int)(q / Tt);
    size_t base = (((size_t)(b * Tt + t) * NH) + h) * Nn + 2 * (size_t)np;
    float ve = (float)xs[base], vo = (float)xs[base + 1];
    float freq = exp2f(-16.f * (float)(2 * np) / (float)Nn) / TWOPI;
    float ph   = (float)t * freq;
    float ang  = (ph - floorf(ph)) * TWOPI;
    float c = cosf(ang), s = sinf(ang);
    qr[base]     = (__bf16)(ve * c - vo * s);
    qr[base + 1] = (__bf16)(vo * c + ve * s);
}

// ---------------------------------------------------------------------------
extern "C" void kernel_launch(void* const* d_in, const int* in_sizes, int n_in,
                              void* d_out, int out_size, void* d_ws, size_t ws_size,
                              hipStream_t stream)
{
    const int*   idx   = (const int*)  d_in[0];
    const float* embed = (const float*)d_in[1];
    const float* enc   = (const float*)d_in[2];
    const float* encv  = (const float*)d_in[3];
    const float* dec   = (const float*)d_in[4];
    const float* lmh   = (const float*)d_in[5];
    float* out = (float*)d_out;

    char* w = (char*)d_ws; size_t off = 0;
    auto take = [&](size_t bytes) -> void* {
        void* p = w + off; off += (bytes + 255) & ~(size_t)255; return p;
    };
    __bf16* enc_t  = (__bf16*)take((size_t)NH * Nn * Dd * 2);   // (h, n, d)
    __bf16* encv_t = (__bf16*)take((size_t)NH * Nn * Dd * 2);   // (h, n, d)
    __bf16* dec_t  = (__bf16*)take((size_t)Dd * NHN * 2);       // (d, h*N+n)
    __bf16* lmh_t  = (__bf16*)take((size_t)256 * Dd * 2);       // (v, d)
    __bf16* xs     = (__bf16*)take((size_t)BT * NHN * 2);       // (b,t,h,n)
    __bf16* qr     = (__bf16*)take((size_t)BT * NHN * 2);       // (b,t,h,n)
    __bf16* sc     = (__bf16*)take((size_t)Bb * NH * Tt * Tt * 2); // (b,h,t,s)
    float*  ykv    = (float*) take((size_t)NH * BT * Dd * 4);   // (h,b,t,d)
    __bf16* ykvln  = (__bf16*)take((size_t)NH * BT * Dd * 2);   // (h,b,t,d)
    float*  x      = (float*) take((size_t)BT * Dd * 4);
    __bf16* xbf    = (__bf16*)take((size_t)BT * Dd * 2);
    __bf16* xT     = (__bf16*)take((size_t)Bb * Dd * Tt * 2);   // (b,d,t)
    float*  ymlp   = (float*) take((size_t)BT * Dd * 4);

    // --- weight prep: fp32 -> bf16 transposed (once per launch) ---
    k_w_transpose<<<dim3((Dd * Nn) / 256, 1, NH), 256, 0, stream>>>(
        enc,  enc_t,  Dd, Nn, (long)Dd * Nn, (long)Nn * Dd);
    k_w_transpose<<<dim3((Dd * Nn) / 256, 1, NH), 256, 0, stream>>>(
        encv, encv_t, Dd, Nn, (long)Dd * Nn, (long)Nn * Dd);
    k_w_transpose<<<dim3(((size_t)NHN * Dd) / 256, 1, 1), 256, 0, stream>>>(
        dec,  dec_t,  NHN, Dd, 0, 0);
    k_w_transpose<<<dim3((Dd * 256) / 256, 1, 1), 256, 0, stream>>>(
        lmh,  lmh_t,  Dd, 256, 0, 0);

    // --- x = LN(embed[idx]) ---
    k_embed_ln<<<BT, 256, 0, stream>>>(idx, embed, x, xbf, xT);

    for (int L = 0; L < 4; ++L) {
        // x_sparse = relu(x @ enc_h)   M=1024 N=8192 K=256, per head
        k_gemm_nt<EP_RELU_BF16><<<dim3(Nn / 256, BT / 64, NH), 256, 0, stream>>>(
            xbf, enc_t, xs, nullptr,
            Dd, Dd, Dd, NHN, 0, NH,
            0, 0, 0, (long)Nn * Dd, 0, (long)Nn, 0, 0);

        // QR = rope(x_sparse)
        k_rope<<<((size_t)BT * NHN / 2) / 256, 256, 0, stream>>>(xs, qr);

        // scores = tril(QR @ QR^T, -1)   M=N=512 K=8192, per (b,h)
        k_gemm_nt<EP_MASK_BF16><<<dim3(Tt / 256, Tt / 64, Bb * NH), 256, 0, stream>>>(
            qr, qr, sc, nullptr,
            Nn, NHN, NHN, Tt, 0, NH,
            (long)Tt * NHN, (long)Nn, (long)Tt * NHN, (long)Nn,
            (long)NH * Tt * Tt, (long)Tt * Tt, 0, 0);

        // yKV = scores @ x   M=512 N=256 K=512, per (b,h), out layout (h,b,t,d)
        k_gemm_nt<EP_F32><<<dim3(Dd / 256, Tt / 64, Bb * NH), 256, 0, stream>>>(
            sc, xT, ykv, nullptr,
            Tt, Tt, Tt, Dd, 0, NH,
            (long)NH * Tt * Tt, (long)Tt * Tt, (long)Dd * Tt, 0,
            (long)Tt * Dd, (long)BT * Dd, 0, 0);

        // LN(yKV) -> bf16
        k_ln_rows<<<NH * BT, 256, 0, stream>>>(ykv, ykvln);

        // xy = x_sparse * relu(yKVln @ encv_h), in place into xs
        k_gemm_nt<EP_GATE_BF16><<<dim3(Nn / 256, BT / 64, NH), 256, 0, stream>>>(
            ykvln, encv_t, xs, xs,
            Dd, Dd, Dd, NHN, NHN, NH,
            0, (long)BT * Dd, 0, (long)Nn * Dd, 0, (long)Nn, 0, (long)Nn);

        // yMLP = xy @ dec   M=1024 N=256 K=32768
        k_gemm_nt<EP_F32><<<dim3(Dd / 256, BT / 64, 1), 256, 0, stream>>>(
            xs, dec_t, ymlp, nullptr,
            NHN, NHN, NHN, Dd, 0, 1,
            0, 0, 0, 0, 0, 0, 0, 0);

        // x = LN(x + yMLP)
        k_resid_ln<<<BT, 256, 0, stream>>>(x, ymlp, xbf, xT);
    }

    // logits = x @ lm_head   M=1024 N=256 K=256
    k_gemm_nt<EP_F32><<<dim3(256 / 256, BT / 64, 1), 256, 0, stream>>>(
        xbf, lmh_t, out, nullptr,
        Dd, Dd, Dd, 256, 0, 1,
        0, 0, 0, 0, 0, 0, 0, 0);
}
